// LiquidTimeConstantCell_9096740733142
// MI455X (gfx1250) — matrix operand order repaired and checked by reading, not compiled, not run on hardware
//
#include <hip/hip_runtime.h>
#include <cmath>

typedef __attribute__((ext_vector_type(16))) _Float16 v16h;
typedef __attribute__((ext_vector_type(2)))  _Float16 v2h;
typedef __attribute__((ext_vector_type(8)))  float    v8f;
typedef __attribute__((ext_vector_type(8)))  int      v8i;

#define RED_BLOCKS  1024
#define RED_THREADS 256
#define WAVES 8              // 8 waves * 16 batch rows = 128 rows per block
#define HDIM 19
#define DDIM 64

// ---------------- ws layout (bytes) ----------------
// [0, 4096)      : 1024 f32 block partial sums
// [4096, 4224)   : 32 f32 scale[j] = dt/tau_dyn[j]   (zero-padded j>=19)
// [4224, 4352)   : 32 f32 bi padded
// [4352, 4480)   : 32 f32 bo padded
// [4480, 12672)  : 8 f16 A-fragments (512 halfs each, [lane][e] order)
//                  fi 0..3 : Wi  (jt = fi>>1, k0 = (fi&1)*32)
//                  fi 4..5 : Wr  (jt)
//                  fi 6..7 : Wo  (jt)
#define WS_SCALE_OFF 4096
#define WS_BIP_OFF   4224
#define WS_BOP_OFF   4352
#define WS_FRAG_OFF  4480

static __device__ __forceinline__ v8f wmma_f16(v16h a, v16h b, v8f c) {
    return __builtin_amdgcn_wmma_f32_16x16x32_f16(false, a, false, b, (short)0, c,
                                                  false, false);
}

static __device__ __forceinline__ float fast_tanh(float v) {
#if __has_builtin(__builtin_amdgcn_tanhf)
    return __builtin_amdgcn_tanhf(v);            // v_tanh_f32 (gfx1250 trans op)
#elif __has_builtin(__builtin_amdgcn_tanh_f32)
    return __builtin_amdgcn_tanh_f32(v);
#else
    float a = fminf(fmaxf(v, -9.0f), 9.0f);      // avoid inf*0 in rcp path
    float e = __expf(2.0f * a);
    return 1.0f - 2.0f * __builtin_amdgcn_rcpf(e + 1.0f);
#endif
}

// load 16 consecutive floats, convert to one f16 x16 fragment
static __device__ __forceinline__ v16h load_row16_f16(const float* p) {
    float4 f0 = *(const float4*)(p);
    float4 f1 = *(const float4*)(p + 4);
    float4 f2 = *(const float4*)(p + 8);
    float4 f3 = *(const float4*)(p + 12);
    float f[16] = {f0.x, f0.y, f0.z, f0.w, f1.x, f1.y, f1.z, f1.w,
                   f2.x, f2.y, f2.z, f2.w, f3.x, f3.y, f3.z, f3.w};
    v16h r;
#pragma unroll
    for (int e = 0; e < 16; e++) r[e] = (_Float16)f[e];
    return r;
}

// Build h^T B-fragment (K = hidden j, N = batch col) from the two C/D-layout
// h tiles, using ds_swizzle SWAPX16 (lane <-> lane^16) — no LDS memory.
static __device__ __forceinline__ v16h build_hb(const v8f& h0, const v8f& h1, int kh) {
    int pk0[4], pk1[4];
#pragma unroll
    for (int d = 0; d < 4; d++) {
        v2h a = {(_Float16)h0[2 * d], (_Float16)h0[2 * d + 1]};
        v2h b = {(_Float16)h1[2 * d], (_Float16)h1[2 * d + 1]};
        pk0[d] = __builtin_bit_cast(int, a);
        pk1[d] = __builtin_bit_cast(int, b);
    }
    int sw0[4], sw1[4];
#pragma unroll
    for (int d = 0; d < 4; d++) {
        sw0[d] = __builtin_amdgcn_ds_swizzle(pk0[d], 0x401f);  // SWAPX16
        sw1[d] = __builtin_amdgcn_ds_swizzle(pk1[d], 0x401f);
    }
    v8i w;
#pragma unroll
    for (int d = 0; d < 4; d++) {
        // kh=0: e=0..7 -> own tile0 (j=0..7);   e=8..15 -> partner tile0 (j=8..15)
        // kh=1: e=0..7 -> partner tile1 (j=16..23); e=8..15 -> own tile1 (j=24..31)
        w[d]     = kh ? sw1[d] : pk0[d];
        w[d + 4] = kh ? pk1[d] : sw0[d];
    }
    return __builtin_bit_cast(v16h, w);
}

// -------- kernel 1: deterministic |x| partial sums --------
__global__ __launch_bounds__(RED_THREADS)
void k_abs_sum(const float* __restrict__ x, float* __restrict__ partials, int n4) {
    __shared__ float s[RED_THREADS];
    const float4* x4 = (const float4*)x;
    float acc = 0.0f;
    for (int i = blockIdx.x * blockDim.x + threadIdx.x; i < n4;
         i += gridDim.x * blockDim.x) {
        float4 v = x4[i];
        acc += fabsf(v.x) + fabsf(v.y) + fabsf(v.z) + fabsf(v.w);
    }
    s[threadIdx.x] = acc;
    __syncthreads();
    for (int off = RED_THREADS / 2; off > 0; off >>= 1) {
        if (threadIdx.x < (unsigned)off) s[threadIdx.x] += s[threadIdx.x + off];
        __syncthreads();
    }
    if (threadIdx.x == 0) partials[blockIdx.x] = s[0];
}

// -------- kernel 2: urgency -> scale / padded biases, pack W into A-fragments --------
__global__ __launch_bounds__(256)
void k_finalize(const float* __restrict__ partials,
                const float* __restrict__ tau, const float* __restrict__ ta,
                const float* __restrict__ bi, const float* __restrict__ bo,
                const float* __restrict__ Wi, const float* __restrict__ Wr,
                const float* __restrict__ Wo,
                float* __restrict__ scaleOut, float* __restrict__ bip,
                float* __restrict__ bop, _Float16* __restrict__ frags,
                long long nElem) {
    __shared__ float s[256];
    __shared__ float urg;
    float acc = 0.0f;
    for (int i = threadIdx.x; i < RED_BLOCKS; i += 256) acc += partials[i];
    s[threadIdx.x] = acc;
    __syncthreads();
    for (int off = 128; off > 0; off >>= 1) {
        if (threadIdx.x < (unsigned)off) s[threadIdx.x] += s[threadIdx.x + off];
        __syncthreads();
    }
    if (threadIdx.x == 0) {
        float mean = s[0] / (float)nElem;
        urg = fmaxf(mean, 0.01f);
    }
    __syncthreads();

    if (threadIdx.x < 32) {
        int j = threadIdx.x;
        float sc = 0.0f, vb = 0.0f, vo = 0.0f;
        if (j < HDIM) {
            float td = tau[j] * (1.0f - ta[j]) + ta[j] / urg;
            td = fminf(fmaxf(td, 0.01f), 10.0f);
            sc = 0.01f / td;                       // dt / tau_dyn
            vb = bi[j];
            vo = bo[j];
        }
        scaleOut[j] = sc;
        bip[j] = vb;
        bop[j] = vo;
    }

    // A-fragment pack: element i = fi*512 + lane*16 + e
    // A layout: row j = jt*16 + (lane&15);  K = k0 + e + (lane>>4)*16
    for (int i = threadIdx.x; i < 8 * 512; i += 256) {
        int fi = i >> 9, w = i & 511, lane = w >> 4, e = w & 15;
        int m = lane & 15, kh = lane >> 4;
        int jt, k0, ld, kmax;
        const float* src;
        if (fi < 4)      { jt = fi >> 1; k0 = (fi & 1) * 32; src = Wi; ld = DDIM; kmax = DDIM; }
        else if (fi < 6) { jt = fi - 4;  k0 = 0;             src = Wr; ld = HDIM; kmax = HDIM; }
        else             { jt = fi - 6;  k0 = 0;             src = Wo; ld = HDIM; kmax = HDIM; }
        int j = jt * 16 + m;
        int K = k0 + e + kh * 16;
        float v = (j < HDIM && K < kmax) ? src[j * ld + K] : 0.0f;
        frags[i] = (_Float16)v;
    }
}

// -------- kernel 3: fused, transposed, LDS-free recurrence --------
__global__ __launch_bounds__(32 * WAVES)
void k_ltc(const float* __restrict__ x, const float* __restrict__ scale,
           const float* __restrict__ bip, const float* __restrict__ bop,
           const _Float16* __restrict__ frags, const int* __restrict__ stepsPtr,
           float* __restrict__ out, long long Bn) {
    const int lane = threadIdx.x & 31;
    const int wave = threadIdx.x >> 5;
    const int b  = lane & 15;   // batch column within tile
    const int kh = lane >> 4;   // lane half
    const long long row0 = ((long long)blockIdx.x * WAVES + wave) * 16;

    const int steps = *stepsPtr;

    // per-row (hidden j) constants; in transposed C/D layout j = jt*16 + kh*8 + r
    float biT[2][8], boT[2][8], sT[2][8];
#pragma unroll
    for (int jt = 0; jt < 2; jt++) {
        int base = jt * 16 + kh * 8;
        float4 a0 = *(const float4*)(bip + base);
        float4 a1 = *(const float4*)(bip + base + 4);
        float4 b0 = *(const float4*)(bop + base);
        float4 b1 = *(const float4*)(bop + base + 4);
        float4 s0 = *(const float4*)(scale + base);
        float4 s1 = *(const float4*)(scale + base + 4);
        float tb[8] = {a0.x, a0.y, a0.z, a0.w, a1.x, a1.y, a1.z, a1.w};
        float to[8] = {b0.x, b0.y, b0.z, b0.w, b1.x, b1.y, b1.z, b1.w};
        float ts[8] = {s0.x, s0.y, s0.z, s0.w, s1.x, s1.y, s1.z, s1.w};
#pragma unroll
        for (int r = 0; r < 8; r++) { biT[jt][r] = tb[r]; boT[jt][r] = to[r]; sT[jt][r] = ts[r]; }
    }

    // constant weight A-fragments (pre-packed f16)
    v16h wiA[2][2], wrA[2], woA[2];
#pragma unroll
    for (int jt = 0; jt < 2; jt++) {
#pragma unroll
        for (int kf = 0; kf < 2; kf++)
            wiA[jt][kf] = *(const v16h*)(frags + (jt * 2 + kf) * 512 + lane * 16);
        wrA[jt] = *(const v16h*)(frags + (4 + jt) * 512 + lane * 16);
        woA[jt] = *(const v16h*)(frags + (6 + jt) * 512 + lane * 16);
    }

    // x^T B-fragments: lane b, K = e + kh*16 + kf*32 -> 16 consecutive floats
    v16h xb[2];
    {
        const float* xr = x + (row0 + b) * DDIM;
#pragma unroll
        for (int kf = 0; kf < 2; kf++) xb[kf] = load_row16_f16(xr + kf * 32 + kh * 16);
    }

    // mapped^T = Wi x^T + bi
    v8f mapC[2], hC[2];
#pragma unroll
    for (int jt = 0; jt < 2; jt++) {
        v8f c;
#pragma unroll
        for (int r = 0; r < 8; r++) c[r] = biT[jt][r];
        c = wmma_f16(wiA[jt][0], xb[0], c);
        c = wmma_f16(wiA[jt][1], xb[1], c);
        mapC[jt] = c;
#pragma unroll
        for (int r = 0; r < 8; r++) hC[jt][r] = 0.0f;
    }

    // recurrence: h += (tanh(mapped + Wr h) - h) * dt/tau_dyn   (all transposed)
    for (int s = 0; s < steps; ++s) {
        v16h hb = build_hb(hC[0], hC[1], kh);
#pragma unroll
        for (int jt = 0; jt < 2; jt++) {
            v8f act = wmma_f16(wrA[jt], hb, mapC[jt]);
#pragma unroll
            for (int r = 0; r < 8; r++) {
                float t = fast_tanh(act[r]);
                hC[jt][r] += (t - hC[jt][r]) * sT[jt][r];
            }
        }
    }

    // out^T = Wo h^T + bo
    v16h hb = build_hb(hC[0], hC[1], kh);
    v8f oD[2];
#pragma unroll
    for (int jt = 0; jt < 2; jt++) {
        v8f c;
#pragma unroll
        for (int r = 0; r < 8; r++) c[r] = boT[jt][r];
        oD[jt] = wmma_f16(woA[jt], hb, c);
    }

    // stores: lane owns batch row (row0 + b), hidden slice j = jt*16 + kh*8 + r
    const long long row = row0 + b;
    float* o0 = out + row * HDIM;                 // h@Wo^T + bo
    float* o1 = out + Bn * HDIM + row * HDIM;     // h
#pragma unroll
    for (int r = 0; r < 8; r++) {
        int j = kh * 8 + r;                       // 0..15, always valid
        o0[j] = oD[0][r];
        o1[j] = hC[0][r];
    }
#pragma unroll
    for (int r = 0; r < 8; r++) {
        int j = 16 + kh * 8 + r;                  // valid only 16..18 (kh=0, r<3)
        if (j < HDIM) { o0[j] = oD[1][r]; o1[j] = hC[1][r]; }
    }
}

extern "C" void kernel_launch(void* const* d_in, const int* in_sizes, int n_in,
                              void* d_out, int out_size, void* d_ws, size_t ws_size,
                              hipStream_t stream) {
    const float* x   = (const float*)d_in[0];
    const float* Wi  = (const float*)d_in[1];
    const float* bi  = (const float*)d_in[2];
    const float* Wr  = (const float*)d_in[3];
    const float* Wo  = (const float*)d_in[4];
    const float* bo  = (const float*)d_in[5];
    const float* tau = (const float*)d_in[6];
    const float* ta  = (const float*)d_in[7];
    const int* steps = (const int*)d_in[8];

    const long long B = (long long)in_sizes[0] / DDIM;   // 524288
    float*    partials = (float*)d_ws;
    float*    scale    = (float*)((char*)d_ws + WS_SCALE_OFF);
    float*    bip      = (float*)((char*)d_ws + WS_BIP_OFF);
    float*    bop      = (float*)((char*)d_ws + WS_BOP_OFF);
    _Float16* frags    = (_Float16*)((char*)d_ws + WS_FRAG_OFF);

    k_abs_sum<<<RED_BLOCKS, RED_THREADS, 0, stream>>>(x, partials,
                                                      (int)(B * DDIM / 4));
    k_finalize<<<1, 256, 0, stream>>>(partials, tau, ta, bi, bo, Wi, Wr, Wo,
                                      scale, bip, bop, frags, B * DDIM);
    k_ltc<<<(int)(B / (16 * WAVES)), 32 * WAVES, 0, stream>>>(
        x, scale, bip, bop, frags, steps, (float*)d_out, B);
}
